// MyMoE_73366631350451
// MI455X (gfx1250) — matrix operand order repaired and checked by reading, not compiled, not run on hardware
//
#include <hip/hip_runtime.h>
#include <hip/hip_bf16.h>
#include <stdint.h>

// Problem constants (match reference)
#define B_  4
#define S_  2048
#define H_  1024
#define E_  8
#define M_  512
#define SM_ 1024
#define G_  64
#define T_  8192   // B*S tokens

typedef __bf16 bf16x16 __attribute__((ext_vector_type(16)));
typedef __bf16 bf16x8  __attribute__((ext_vector_type(8)));
typedef float  v8f     __attribute__((ext_vector_type(8)));
typedef unsigned int u32x4 __attribute__((ext_vector_type(4)));
typedef int i32x4 __attribute__((ext_vector_type(4)));
typedef int i32x8 __attribute__((ext_vector_type(8)));

#if __has_include(<hip/amd_detail/amd_gfx1250_TDM.h>)
#define TDM_6ARG 1
#else
#define TDM_6ARG 0
#endif

// fp32 -> bf16 round-to-nearest-even
__device__ __forceinline__ unsigned short f2bf(float f){
  union { float f; unsigned u; } v; v.f = f;
  unsigned u = v.u;
  u += 0x7FFFu + ((u >> 16) & 1u);
  return (unsigned short)(u >> 16);
}

__device__ __forceinline__ bf16x16 frag_pair(const unsigned short* p0, const unsigned short* p1){
  bf16x8 lo = *reinterpret_cast<const bf16x8*>(p0);
  bf16x8 hi = *reinterpret_cast<const bf16x8*>(p1);
  bf16x16 r;
  #pragma unroll
  for (int i = 0; i < 8; ++i){ r[i] = lo[i]; r[i+8] = hi[i]; }
  return r;
}

// A fragment: 16xK row-major tile (stride elems). Dense 16-bit A layout:
// lanes 0-15 (row=lane): K = k0+0..7 and k0+16..23; lanes 16-31: +8.
__device__ __forceinline__ bf16x16 ldA(const unsigned short* base, int stride, int k0){
  int lane = threadIdx.x & 31;
  int half = lane >> 4;
  int m    = lane & 15;
  const unsigned short* p0 = base + (size_t)m*stride + k0 + half*8;
  return frag_pair(p0, p0 + 16);
}

// B fragment from N-major (pre-transposed) weights Bt[n][k]:
// lane n = n0 + (lane&15); lanes 0-15 hold K=k0+0..7 and k0+16..23; lanes 16-31: +8.
__device__ __forceinline__ bf16x16 ldB(const unsigned short* bt, int n0, int strideK, int k0){
  int lane = threadIdx.x & 31;
  int half = lane >> 4;
  int n    = n0 + (lane & 15);
  const unsigned short* p = bt + (size_t)n*strideK + k0 + half*8;
  return frag_pair(p, p + 16);
}

__device__ __forceinline__ v8f wmma_bf16(bf16x16 a, bf16x16 b, v8f c){
  return __builtin_amdgcn_wmma_f32_16x16x32_bf16(false, a, false, b, (short)0, c, false, false);
}

// ---------------------------------------------------------------- init
__global__ __launch_bounds__(256) void k_init(float* __restrict__ y_acc, int* __restrict__ cnt){
  size_t i = (size_t)blockIdx.x*blockDim.x + threadIdx.x;
  if (i < E_) cnt[i] = 0;
  size_t n = (size_t)T_*H_;
  size_t stride = (size_t)gridDim.x*blockDim.x;
  for (size_t j = i; j < n; j += stride) y_acc[j] = 0.f;
}

// ---------------------------------------------------------------- fp32 -> bf16
__global__ __launch_bounds__(256) void k_cvt(const float* __restrict__ x,
                                             unsigned short* __restrict__ xb, size_t n){
  size_t stride = (size_t)gridDim.x*blockDim.x;
  for (size_t i = (size_t)blockIdx.x*blockDim.x + threadIdx.x; i < n; i += stride)
    xb[i] = f2bf(x[i]);
}

// ------------------------------------------------- transpose+convert (RxC -> CxR bf16)
__global__ __launch_bounds__(256) void k_tr(const float* __restrict__ src,
                                            unsigned short* __restrict__ dst, int R, int C){
  __shared__ unsigned short tile[32][33];
  const float*    s = src + (size_t)blockIdx.z*R*C;
  unsigned short* d = dst + (size_t)blockIdx.z*R*C;
  int tx = threadIdx.x, ty = threadIdx.y;
  int r0 = blockIdx.y*32, c0 = blockIdx.x*32;
  #pragma unroll
  for (int i = 0; i < 4; ++i){
    int r = r0 + ty + i*8;
    tile[ty + i*8][tx] = f2bf(s[(size_t)r*C + c0 + tx]);
  }
  __syncthreads();
  #pragma unroll
  for (int i = 0; i < 4; ++i){
    int c = c0 + ty + i*8;
    d[(size_t)c*R + r0 + tx] = tile[tx][ty + i*8];
  }
}

// ---------------------------------------------------------------- router: top-2 of 8
__global__ __launch_bounds__(256) void k_router(
    const float* __restrict__ x, const float* __restrict__ g,
    const float* __restrict__ wt, const float* __restrict__ wg,
    int* __restrict__ tokL, float* __restrict__ wgtL, int* __restrict__ cnt)
{
  int wave = threadIdx.x >> 5, lane = threadIdx.x & 31;
  int t = blockIdx.x*8 + wave;
  float p[E_];
  #pragma unroll
  for (int e = 0; e < E_; ++e) p[e] = 0.f;
  const float* xr = x + (size_t)t*H_;
  for (int h = lane; h < H_; h += 32){
    float xv = xr[h];
    #pragma unroll
    for (int e = 0; e < E_; ++e) p[e] += xv * wt[e*H_ + h];
  }
  #pragma unroll
  for (int m = 16; m >= 1; m >>= 1){
    #pragma unroll
    for (int e = 0; e < E_; ++e) p[e] += __shfl_xor(p[e], m, 32);
  }
  int b = t >> 11;  // token -> batch (S=2048)
  float l[E_];
  #pragma unroll
  for (int e = 0; e < E_; ++e){
    float lg = 0.f;
    for (int j = 0; j < G_; ++j) lg += g[b*G_ + j]*wg[e*G_ + j];
    l[e] = 0.5f*(p[e] + lg);          // GE_TEMP = 1 -> (lh + lg)/2
  }
  float mx = l[0];
  #pragma unroll
  for (int e = 1; e < E_; ++e) mx = fmaxf(mx, l[e]);
  float s[E_]; float se = 0.f;
  #pragma unroll
  for (int e = 0; e < E_; ++e){ s[e] = __expf(l[e]-mx); se += s[e]; }
  #pragma unroll
  for (int e = 0; e < E_; ++e) s[e] /= se;
  int i0 = 0;
  #pragma unroll
  for (int e = 1; e < E_; ++e) if (s[e] > s[i0]) i0 = e;
  int i1 = (i0 == 0) ? 1 : 0;
  #pragma unroll
  for (int e = 0; e < E_; ++e) if (e != i0 && s[e] > s[i1]) i1 = e;
  float inv = 1.f/(s[i0]+s[i1]+1e-20f);
  if (lane == 0){
    int p0 = atomicAdd(&cnt[i0], 1);
    tokL[i0*T_ + p0] = t; wgtL[i0*T_ + p0] = s[i0]*inv;
    int p1 = atomicAdd(&cnt[i1], 1);
    tokL[i1*T_ + p1] = t; wgtL[i1*T_ + p1] = s[i1]*inv;
  }
}

// ------------------------------------------ routed expert FFN over gathered 16-token tiles
__global__ __launch_bounds__(256) void k_expert(
    const unsigned short* __restrict__ xb,
    const unsigned short* __restrict__ wgT,   // (E, M, H) bf16, N-major
    const unsigned short* __restrict__ wuT,   // (E, M, H)
    const unsigned short* __restrict__ wdT,   // (E, H, M)
    const int* __restrict__ tokL, const float* __restrict__ wgtL,
    const int* __restrict__ cnt, float* __restrict__ y_acc)
{
  int e    = blockIdx.x >> 9;       // T_/16 = 512 tiles per expert
  int tile = blockIdx.x & 511;
  int c = cnt[e];
  if (tile*16 >= c) return;

  __shared__ unsigned short s_a[16*H_];   // 32 KB: gathered activations (bf16)
  __shared__ unsigned short s_h[16*M_];   // 16 KB: gate*up intermediate (bf16)
  __shared__ int   s_tok[16];
  __shared__ float s_w[16];

  int tid = threadIdx.x;
  if (tid < 16){
    int idx = tile*16 + tid;
    if (idx < c){ s_tok[tid] = tokL[e*T_ + idx]; s_w[tid] = wgtL[e*T_ + idx]; }
    else        { s_tok[tid] = 0;                s_w[tid] = 0.f; }   // pad rows -> weight 0
  }

#if __has_builtin(__builtin_amdgcn_tensor_load_to_lds)
  // --- Tensor Data Mover gather: one TDM op pulls 16 (possibly scattered) token
  // rows of 1024 bf16 each from global xb into LDS, tracked by TENSORcnt.
  if (tid < 32){   // wave 0 issues the DMA (EXEC-independent, per-wave op)
    unsigned idx16[16];
    #pragma unroll
    for (int i = 0; i < 16; ++i){
      int ii = tile*16 + i;                        // uniform -> scalar loads
      idx16[i] = (ii < c) ? (unsigned)tokL[e*T_ + ii] : 0u;
    }
    unsigned lds_lo = (unsigned)(uintptr_t)(void*)s_a;        // LDS byte offset
    unsigned long long ga = (unsigned long long)(uintptr_t)xb;
    u32x4 g0;
    g0[0] = (1u << 31) | 1u;                       // gather_mode=1, 16-bit idx, count=1
    g0[1] = lds_lo;                                // lds_addr
    g0[2] = (unsigned)(ga & 0xFFFFFFFFu);          // global_addr[31:0]
    g0[3] = (unsigned)((ga >> 32) & 0x01FFFFFFu) | (2u << 30);  // addr[56:32], type=2
    i32x8 g1;
    g1[0] = 0x00010000;                 // workgroup_mask=0, data_size=1 (2B)
    g1[1] = (int)(1024u << 16);         // tensor_dim0 = 1024 (low 16 at bit 48)
    g1[2] = (int)(8192u << 16);         // tensor_dim1 = 8192 (low 16 at bit 80)
    g1[3] = (int)(1024u << 16);         // tile_dim0 = 1024 (bits 127:112)
    g1[4] = 16;                         // tile_dim1 = #indices = 16
    g1[5] = 1024;                       // tensor_dim0_stride = 1024 elems
    g1[6] = 0;
    g1[7] = 0;
    i32x4 g2, g3;
    #pragma unroll
    for (int j = 0; j < 4; ++j){
      g2[j] = (int)(idx16[2*j]     | (idx16[2*j + 1] << 16));
      g3[j] = (int)(idx16[8 + 2*j] | (idx16[8 + 2*j + 1] << 16));
    }
#if TDM_6ARG
    i32x8 gz = {};
    __builtin_amdgcn_tensor_load_to_lds(g0, g1, g2, g3, gz, 0);
#else
    __builtin_amdgcn_tensor_load_to_lds(g0, g1, g2, g3, 0);
#endif
    __builtin_amdgcn_s_wait_tensorcnt(0);
  }
  __syncthreads();
#else
  __syncthreads();
  // fallback: vector gather, 16B chunks (128 chunks per 2048B row)
  for (int i = tid; i < 16*128; i += 256){
    int m = i >> 7, ch = i & 127;
    reinterpret_cast<float4*>(s_a)[m*128 + ch] =
      reinterpret_cast<const float4*>(xb)[(size_t)s_tok[m]*128 + ch];
  }
  __syncthreads();
#endif

  int wave = tid >> 5, lane = tid & 31, half = lane >> 4, nn = lane & 15;
  const unsigned short* Bg = wgT + (size_t)e*M_*H_;
  const unsigned short* Bu = wuT + (size_t)e*M_*H_;

  // GEMM1: (16 x 1024) @ (1024 x 512) gate & up. k-outer; per k-step, batch-load
  // all 8 B fragments into distinct registers (one clause burst, staged waits),
  // then the 8-WMMA burst off a single A fragment.
  v8f accG[4], accU[4];
  #pragma unroll
  for (int nt = 0; nt < 4; ++nt){ accG[nt] = {}; accU[nt] = {}; }
  for (int k0 = 0; k0 < H_; k0 += 32){
    bf16x16 a = ldA(s_a, H_, k0);
    bf16x16 bG[4], bU[4];
    #pragma unroll
    for (int nt = 0; nt < 4; ++nt){
      int n0 = wave*64 + nt*16;
      bG[nt] = ldB(Bg, n0, H_, k0);
      bU[nt] = ldB(Bu, n0, H_, k0);
    }
    #pragma unroll
    for (int nt = 0; nt < 4; ++nt){
      accG[nt] = wmma_bf16(a, bG[nt], accG[nt]);
      accU[nt] = wmma_bf16(a, bU[nt], accU[nt]);
    }
  }
  #pragma unroll
  for (int nt = 0; nt < 4; ++nt){
    int n0 = wave*64 + nt*16;
    #pragma unroll
    for (int r = 0; r < 8; ++r){
      float gv = accG[nt][r], uv = accU[nt][r];
      float hv = (gv >= 0.f ? gv : 0.01f*gv) * uv;   // leaky(gate) * up
      s_h[(r + half*8)*M_ + n0 + nn] = f2bf(hv);
    }
  }
  __syncthreads();

  // GEMM2: (16 x 512) @ (512 x 1024). Same batched-B structure.
  const unsigned short* Bd = wdT + (size_t)e*H_*M_;
  v8f acc[8];
  #pragma unroll
  for (int nt = 0; nt < 8; ++nt) acc[nt] = {};
  for (int k0 = 0; k0 < M_; k0 += 32){
    bf16x16 a = ldA(s_h, M_, k0);
    bf16x16 b[8];
    #pragma unroll
    for (int nt = 0; nt < 8; ++nt)
      b[nt] = ldB(Bd, wave*128 + nt*16, M_, k0);
    #pragma unroll
    for (int nt = 0; nt < 8; ++nt)
      acc[nt] = wmma_bf16(a, b[nt], acc[nt]);
  }
  #pragma unroll
  for (int nt = 0; nt < 8; ++nt){
    int n = wave*128 + nt*16 + nn;
    #pragma unroll
    for (int r = 0; r < 8; ++r){
      int m = r + half*8;
      atomicAdd(&y_acc[(size_t)s_tok[m]*H_ + n], acc[nt][r]*s_w[m]);
    }
  }
}

// ------------------------------------------ shared expert + final add (out = y_acc + p)
__global__ __launch_bounds__(256) void k_shared(
    const unsigned short* __restrict__ xb,
    const unsigned short* __restrict__ sgT,   // (SM, H) bf16, N-major
    const unsigned short* __restrict__ suT,   // (SM, H)
    const unsigned short* __restrict__ sdT,   // (H, SM)
    const float* __restrict__ y_acc, float* __restrict__ out)
{
  int t0 = blockIdx.x * 16;                   // contiguous 16-token tile
  __shared__ unsigned short s_h[16*SM_];      // 32 KB intermediate
  int tid = threadIdx.x;
  int wave = tid >> 5, lane = tid & 31, half = lane >> 4, nn = lane & 15;
  const unsigned short* A = xb + (size_t)t0*H_;   // rows contiguous; WGP$/L2 resident

  // GEMM1: (16 x 1024) @ (1024 x 1024) gate & up, two 64-column half-blocks.
  #pragma unroll
  for (int hb = 0; hb < 2; ++hb){
    v8f accG[4], accU[4];
    #pragma unroll
    for (int nt = 0; nt < 4; ++nt){ accG[nt] = {}; accU[nt] = {}; }
    for (int k0 = 0; k0 < H_; k0 += 32){
      bf16x16 a = ldA(A, H_, k0);
      bf16x16 bG[4], bU[4];
      #pragma unroll
      for (int nt = 0; nt < 4; ++nt){
        int n0 = wave*128 + hb*64 + nt*16;
        bG[nt] = ldB(sgT, n0, H_, k0);
        bU[nt] = ldB(suT, n0, H_, k0);
      }
      #pragma unroll
      for (int nt = 0; nt < 4; ++nt){
        accG[nt] = wmma_bf16(a, bG[nt], accG[nt]);
        accU[nt] = wmma_bf16(a, bU[nt], accU[nt]);
      }
    }
    #pragma unroll
    for (int nt = 0; nt < 4; ++nt){
      int n0 = wave*128 + hb*64 + nt*16;
      #pragma unroll
      for (int r = 0; r < 8; ++r){
        float gv = accG[nt][r], uv = accU[nt][r];
        s_h[(r + half*8)*SM_ + n0 + nn] = f2bf((gv >= 0.f ? gv : 0.01f*gv)*uv);
      }
    }
  }
  __syncthreads();

  // GEMM2: (16 x 1024) @ (1024 x 1024) + residual add from y_acc.
  v8f acc[8];
  #pragma unroll
  for (int nt = 0; nt < 8; ++nt) acc[nt] = {};
  for (int k0 = 0; k0 < SM_; k0 += 32){
    bf16x16 a = ldA(s_h, SM_, k0);
    bf16x16 b[8];
    #pragma unroll
    for (int nt = 0; nt < 8; ++nt)
      b[nt] = ldB(sdT, wave*128 + nt*16, SM_, k0);
    #pragma unroll
    for (int nt = 0; nt < 8; ++nt)
      acc[nt] = wmma_bf16(a, b[nt], acc[nt]);
  }
  #pragma unroll
  for (int nt = 0; nt < 8; ++nt){
    int n = wave*128 + nt*16 + nn;
    #pragma unroll
    for (int r = 0; r < 8; ++r){
      int m = r + half*8;
      size_t o = (size_t)(t0 + m)*H_ + n;
      out[o] = y_acc[o] + acc[nt][r];
    }
  }
}

extern "C" void kernel_launch(void* const* d_in, const int* in_sizes, int n_in,
                              void* d_out, int out_size, void* d_ws, size_t ws_size,
                              hipStream_t stream)
{
  (void)in_sizes; (void)n_in; (void)out_size; (void)ws_size;
  const float* x       = (const float*)d_in[0];
  const float* g       = (const float*)d_in[1];
  const float* wt      = (const float*)d_in[2];
  const float* wg      = (const float*)d_in[3];
  const float* w_gate  = (const float*)d_in[4];
  const float* w_up    = (const float*)d_in[5];
  const float* w_down  = (const float*)d_in[6];
  const float* sw_gate = (const float*)d_in[7];
  const float* sw_up   = (const float*)d_in[8];
  const float* sw_down = (const float*)d_in[9];
  float* out = (float*)d_out;

  char* ws = (char*)d_ws;
  size_t o = 0;
  unsigned short* xb  = (unsigned short*)(ws + o); o += (size_t)T_*H_*2;     // 16 MB
  unsigned short* wgT = (unsigned short*)(ws + o); o += (size_t)E_*H_*M_*2;  //  8 MB
  unsigned short* wuT = (unsigned short*)(ws + o); o += (size_t)E_*H_*M_*2;  //  8 MB
  unsigned short* wdT = (unsigned short*)(ws + o); o += (size_t)E_*M_*H_*2;  //  8 MB
  unsigned short* sgT = (unsigned short*)(ws + o); o += (size_t)H_*SM_*2;    //  2 MB
  unsigned short* suT = (unsigned short*)(ws + o); o += (size_t)H_*SM_*2;    //  2 MB
  unsigned short* sdT = (unsigned short*)(ws + o); o += (size_t)SM_*H_*2;    //  2 MB
  float* y_acc = (float*)(ws + o); o += (size_t)T_*H_*4;                     // 32 MB
  int*   tokL  = (int*)(ws + o);   o += (size_t)E_*T_*4;
  float* wgtL  = (float*)(ws + o); o += (size_t)E_*T_*4;
  int*   cnt   = (int*)(ws + o);   o += 256;                                 // ~78.5 MiB total

  dim3 trB(32, 8, 1);
  k_init  <<<2048, 256, 0, stream>>>(y_acc, cnt);
  k_cvt   <<<4096, 256, 0, stream>>>(x, xb, (size_t)T_*H_);
  k_tr    <<<dim3(M_/32,  H_/32,  E_), trB, 0, stream>>>(w_gate,  wgT, H_,  M_);
  k_tr    <<<dim3(M_/32,  H_/32,  E_), trB, 0, stream>>>(w_up,    wuT, H_,  M_);
  k_tr    <<<dim3(H_/32,  M_/32,  E_), trB, 0, stream>>>(w_down,  wdT, M_,  H_);
  k_tr    <<<dim3(SM_/32, H_/32,  1),  trB, 0, stream>>>(sw_gate, sgT, H_,  SM_);
  k_tr    <<<dim3(SM_/32, H_/32,  1),  trB, 0, stream>>>(sw_up,   suT, H_,  SM_);
  k_tr    <<<dim3(H_/32,  SM_/32, 1),  trB, 0, stream>>>(sw_down, sdT, SM_, H_);
  k_router<<<T_/8,  256, 0, stream>>>(x, g, wt, wg, tokL, wgtL, cnt);
  k_expert<<<E_*(T_/16), 256, 0, stream>>>(xb, wgT, wuT, wdT, tokL, wgtL, cnt, y_acc);
  k_shared<<<T_/16, 256, 0, stream>>>(xb, sgT, suT, sdT, y_acc, out);
}
